// CausalSelfAttention_85624468013743
// MI455X (gfx1250) — compile-verified, matrix operand order checked by reading, not run on hardware
//
#include <hip/hip_runtime.h>
#include <hip/hip_bf16.h>

typedef __bf16         v16bf  __attribute__((ext_vector_type(16)));
typedef float          v8f    __attribute__((ext_vector_type(8)));
typedef float          vf4    __attribute__((ext_vector_type(4)));
typedef unsigned short u16v8  __attribute__((ext_vector_type(8)));
typedef unsigned short u16v16 __attribute__((ext_vector_type(16)));
typedef unsigned int   v4u    __attribute__((ext_vector_type(4)));
typedef int            v8i    __attribute__((ext_vector_type(8)));
typedef int            v4i    __attribute__((ext_vector_type(4)));

constexpr int BATCH = 4, T = 2048, C = 1024, NH = 16, HD = 64;

__device__ __forceinline__ unsigned short f32_to_bf16(float f) {
  unsigned int u = __builtin_bit_cast(unsigned int, f);
  u += 0x7FFFu + ((u >> 16) & 1u);          // round-to-nearest-even
  return (unsigned short)(u >> 16);
}

// Load one 16x32 bf16 WMMA operand fragment from row-major [row][k] storage
// (LDS or global). CDNA5 16-bit A/B layout: lanes 0-15 hold K {0..7,16..23},
// lanes 16-31 hold K {8..15,24..31} of their row/column.
__device__ __forceinline__ v16bf load_frag_rm(const unsigned short* base, int stride,
                                              int row, int kbase, int laneHi) {
  const unsigned short* p = base + (size_t)row * stride + kbase + laneHi * 8;
  u16v8 lo = *(const u16v8*)(p);
  u16v8 hi = *(const u16v8*)(p + 16);
  u16v16 t = __builtin_shufflevector(lo, hi, 0,1,2,3,4,5,6,7,8,9,10,11,12,13,14,15);
  return __builtin_bit_cast(v16bf, t);
}

__device__ __forceinline__ v8f wmma_bf16(v16bf a, v16bf b, v8f c) {
  return __builtin_amdgcn_wmma_f32_16x16x32_bf16(false, a, false, b, (short)0, c,
                                                 false, false);
}

// ---------------------------------------------------------------------------
// TDM: DMA a 2-D bf16 tile (tile0 x tile1 elements, row length tile0,
// row stride stride0 elements) from global memory into LDS. D# built per
// CDNA5 ISA 8.3/8.4: group0 = {count=1, lds_addr, global_addr, type=2},
// group1 = {data_size=2B, tensor dims, tile dims, dim0 stride}.
// Toolchain uses the 6-arg builtin: (g0, g1, g2, g3, g4, cpol).
// ---------------------------------------------------------------------------
__device__ __forceinline__ void tdm_load_tile_2d(const unsigned short* gbase,
                                                 unsigned lds_off, int tdim0,
                                                 int tdim1, int tile0, int tile1,
                                                 int stride0) {
  unsigned long long ga = (unsigned long long)(uintptr_t)gbase;
  v4u g0;
  g0[0] = 1u;                                    // count=1, user descriptor
  g0[1] = lds_off;                               // LDS byte address
  g0[2] = (unsigned)ga;                          // global_addr[31:0]
  g0[3] = (unsigned)(ga >> 32) | 0x80000000u;    // global_addr[56:32] | type=2
  v8i g1;
  g1[0] = 1 << 16;                               // data_size = 1 -> 2 bytes
  g1[1] = (tdim0 & 0xFFFF) << 16;                // tensor_dim0[15:0]
  g1[2] = ((unsigned)tdim0 >> 16) | ((tdim1 & 0xFFFF) << 16);
  g1[3] = ((unsigned)tdim1 >> 16) | (tile0 << 16);  // tile_dim0
  g1[4] = tile1 & 0xFFFF;                        // tile_dim1 (tile_dim2 = 0)
  g1[5] = stride0;                               // tensor_dim0_stride[31:0]
  g1[6] = 0;
  g1[7] = 0;
  v4i z4 = {0, 0, 0, 0};
  v8i z8 = {0, 0, 0, 0, 0, 0, 0, 0};
  __builtin_amdgcn_tensor_load_to_lds(g0, g1, z4, z4, z8, 0);
}

// ---------------------------------------------------------------------------
// One-time pre-passes (memory-bound, ~3us total at 23.3 TB/s):
//   f32 -> bf16 copy, and f32 [K][N] -> bf16 transposed [N][K] weights.
// ---------------------------------------------------------------------------
__global__ __launch_bounds__(256) void convert_bf16_kernel(
    const float* __restrict__ in, unsigned short* __restrict__ out) {
  const size_t i = ((size_t)blockIdx.x * 256 + threadIdx.x) * 8;
  vf4 f0 = *(const vf4*)(in + i);
  vf4 f1 = *(const vf4*)(in + i + 4);
  u16v8 v;
  v[0] = f32_to_bf16(f0[0]); v[1] = f32_to_bf16(f0[1]);
  v[2] = f32_to_bf16(f0[2]); v[3] = f32_to_bf16(f0[3]);
  v[4] = f32_to_bf16(f1[0]); v[5] = f32_to_bf16(f1[1]);
  v[6] = f32_to_bf16(f1[2]); v[7] = f32_to_bf16(f1[3]);
  *(u16v8*)(out + i) = v;
}

__global__ __launch_bounds__(256) void transpose_bf16_kernel(
    const float* __restrict__ w, unsigned short* __restrict__ wt, int N) {
  const size_t i = (size_t)blockIdx.x * 256 + threadIdx.x;  // over N*K, K=1024
  const int n = (int)(i >> 10);
  const int k = (int)(i & 1023);
  wt[i] = f32_to_bf16(w[(size_t)k * N + n]);
}

// ---------------------------------------------------------------------------
// Tiled GEMM: C[M,N] = A[M,K] * Bt[N,K]^T, all-bf16 operands, f32 accumulate.
// WG tile 128x128x32, TDM-staged LDS tiles, 8 waves each owning 32x64
// (2x4 WMMA tiles -> 8 v_wmma per K-step, 6 fragment loads).
// MODE 1: scatter bf16 qkv[3][B][H][T][D];  MODE 0: f32 row-major outF.
// ---------------------------------------------------------------------------
template <int MODE>
__global__ __launch_bounds__(256) void gemm_wmma_kernel(
    const unsigned short* __restrict__ A, const unsigned short* __restrict__ Bt,
    float* __restrict__ outF, unsigned short* __restrict__ outQKV,
    int M, int N, int K) {
  constexpr int BM = 128, BN = 128, BK = 32;
  __shared__ __align__(16) unsigned short As[BM][BK];  // [m][k]
  __shared__ __align__(16) unsigned short Bs[BN][BK];  // [n][k]

  const int tid    = threadIdx.x;
  const int wave   = tid >> 5;
  const int lane   = tid & 31;
  const int lane15 = lane & 15;
  const int laneHi = (lane >> 4) & 1;
  const int wm = wave >> 1, wn = wave & 1;             // 4x2 wave grid
  const int blockM = blockIdx.y * BM;
  const int blockN = blockIdx.x * BN;

  const unsigned ldsA = (unsigned)(uintptr_t)&As[0][0];
  const unsigned ldsB = (unsigned)(uintptr_t)&Bs[0][0];

  const v8f zero8 = {0.f, 0.f, 0.f, 0.f, 0.f, 0.f, 0.f, 0.f};
  v8f acc[2][4] = {{zero8, zero8, zero8, zero8}, {zero8, zero8, zero8, zero8}};

  for (int k0 = 0; k0 < K; k0 += BK) {
    __syncthreads();                     // LDS reads of previous step done
    if (wave == 0) {                     // TDM: EXEC-independent, per-wave issue
      tdm_load_tile_2d(A  + (size_t)blockM * K + k0, ldsA, K, M, BK, BM, K);
      tdm_load_tile_2d(Bt + (size_t)blockN * K + k0, ldsB, K, N, BK, BN, K);
      __builtin_amdgcn_s_wait_tensorcnt(0);
    }
    __syncthreads();                     // tiles visible to all waves

    v16bf a0 = load_frag_rm(&As[0][0], BK, wm * 32 + lane15,      0, laneHi);
    v16bf a1 = load_frag_rm(&As[0][0], BK, wm * 32 + 16 + lane15, 0, laneHi);
#pragma unroll
    for (int nt = 0; nt < 4; ++nt) {
      v16bf b = load_frag_rm(&Bs[0][0], BK, wn * 64 + nt * 16 + lane15, 0, laneHi);
      acc[0][nt] = wmma_bf16(a0, b, acc[0][nt]);
      acc[1][nt] = wmma_bf16(a1, b, acc[1][nt]);
    }
  }

  // epilogue: C-layout VGPR r -> (M = base+r [+8 hi half], N = lane15)
#pragma unroll
  for (int mt = 0; mt < 2; ++mt)
#pragma unroll
    for (int nt = 0; nt < 4; ++nt) {
      const int mbase = blockM + wm * 32 + mt * 16 + laneHi * 8;
      const int ncol  = blockN + wn * 64 + nt * 16 + lane15;
#pragma unroll
      for (int r = 0; r < 8; ++r) {
        const int mrow = mbase + r;
        const float v  = acc[mt][nt][r];
        if constexpr (MODE == 0) {
          outF[(size_t)mrow * N + ncol] = v;
        } else {
          const int which = ncol >> 10;      // 0=q 1=k 2=v
          const int c = ncol & 1023;
          const int h = c >> 6;
          const int d = c & 63;
          const int b = mrow >> 11;          // T = 2048
          const int t = mrow & 2047;
          const size_t idx = ((((size_t)which * BATCH + b) * NH + h) * T + t) * HD + d;
          outQKV[idx] = f32_to_bf16(v);
        }
      }
    }
}

// ---------------------------------------------------------------------------
// Flash attention: one wave per 16-row q tile of one (b,h).
// S^T = K_tile * Q^T (M=key, N=q) -> its f32 C-layout is lane-identical to
// the bf16 A-fragment layout of P for y += P * V (no cross-lane movement).
// ---------------------------------------------------------------------------
__global__ __launch_bounds__(256) void attn_wmma_kernel(
    const unsigned short* __restrict__ qkv,  // [3][B][H][T][D] bf16
    unsigned short* __restrict__ yout) {     // [B][T][C] bf16
  const int wave   = threadIdx.x >> 5;
  const int lane   = threadIdx.x & 31;
  const int lane15 = lane & 15;
  const int laneHi = (lane >> 4) & 1;

  const int tile = blockIdx.x * 8 + wave;    // 8192 tiles total
  const int qt   = tile & 127;               // T/16 = 128 q tiles per head
  const int bh   = tile >> 7;                // 0..63
  const int q0   = qt * 16;

  const size_t hs = (size_t)T * HD;
  const unsigned short* Qh = qkv + (size_t)bh * hs;
  const unsigned short* Kh = qkv + ((size_t)BATCH * NH + bh) * hs;
  const unsigned short* Vh = qkv + ((size_t)2 * BATCH * NH + bh) * hs;

  // Q as B-operand of S^T (column n = q, K-dim = d), loaded once: 2 d-steps.
  v16bf qf[2];
#pragma unroll
  for (int s = 0; s < 2; ++s)
    qf[s] = load_frag_rm(Qh + (size_t)q0 * HD, HD, lane15, s * 32, laneHi);

  const v8f zero8 = {0.f, 0.f, 0.f, 0.f, 0.f, 0.f, 0.f, 0.f};
  v8f y[4] = {zero8, zero8, zero8, zero8};
  float m = -3.0e38f, l = 0.f;
  const int qglob = q0 + lane15;             // this lane's q row
  const int kend  = q0 + 16;                 // causal: skip tiles past diagonal

  for (int kb = 0; kb < kend; kb += 32) {
    if (kb + 32 < kend) {                    // pull next K/V tiles toward L2/L0
      __builtin_prefetch(Kh + (size_t)(kb + 32) * HD, 0, 1);
      __builtin_prefetch(Vh + (size_t)(kb + 32) * HD, 0, 1);
    }
    // ---- S^T: two 16-key subtiles, K-dim = D = 64 (2 WMMA each) ----
    v8f st[2] = {zero8, zero8};
#pragma unroll
    for (int sub = 0; sub < 2; ++sub)
#pragma unroll
      for (int s = 0; s < 2; ++s) {
        v16bf kf = load_frag_rm(Kh + (size_t)(kb + sub * 16) * HD, HD,
                                lane15, s * 32, laneHi);
        st[sub] = wmma_bf16(kf, qf[s], st[sub]);
      }

    // ---- scale, causal mask, online softmax (stats per lane = per q) ----
    float mloc = -3.0e38f;
#pragma unroll
    for (int sub = 0; sub < 2; ++sub)
#pragma unroll
      for (int r = 0; r < 8; ++r) {
        float sv = st[sub][r] * 0.125f;      // 1/sqrt(64)
        const int key = kb + sub * 16 + r + laneHi * 8;
        sv = (key <= qglob) ? sv : -3.0e38f;
        st[sub][r] = sv;
        mloc = fmaxf(mloc, sv);
      }
    mloc = fmaxf(mloc, __shfl_xor(mloc, 16, 32));   // join key halves of row
    const float mnew  = fmaxf(m, mloc);
    const float scale = __expf(m - mnew);

    float psum = 0.f;
    u16v16 pf;
#pragma unroll
    for (int sub = 0; sub < 2; ++sub)
#pragma unroll
      for (int r = 0; r < 8; ++r) {
        const float p = __expf(st[sub][r] - mnew);
        psum += p;
        pf[sub * 8 + r] = f32_to_bf16(p);    // exact P A-fragment slot
      }
    psum += __shfl_xor(psum, 16, 32);
    l = l * scale + psum;
    m = mnew;

    // ---- rescale y accumulators (broadcast per-row scale via bpermute) ----
    float srow[8];
#pragma unroll
    for (int r = 0; r < 8; ++r)
      srow[r] = __shfl(scale, (lane < 16) ? r : (r + 8), 32);
#pragma unroll
    for (int nt = 0; nt < 4; ++nt)
#pragma unroll
      for (int r = 0; r < 8; ++r) y[nt][r] *= srow[r];

    // ---- y += P * V : 4 d-tiles, one K=32 step ----
    const v16bf pA = __builtin_bit_cast(v16bf, pf);
#pragma unroll
    for (int nt = 0; nt < 4; ++nt) {
      u16v16 vv;
#pragma unroll
      for (int e = 0; e < 16; ++e) {
        const int key = kb + (e & 7) + ((e >> 3) << 4) + laneHi * 8;
        vv[e] = Vh[(size_t)key * HD + nt * 16 + lane15];
      }
      y[nt] = wmma_bf16(pA, __builtin_bit_cast(v16bf, vv), y[nt]);
    }
  }

  // ---- normalize and store y as bf16 into [B][T][C] ----
  const float linv = 1.0f / l;
  float lrow[8];
#pragma unroll
  for (int r = 0; r < 8; ++r)
    lrow[r] = __shfl(linv, (lane < 16) ? r : (r + 8), 32);

  const int b = bh >> 4, h = bh & 15;
  unsigned short* yo = yout + (size_t)b * T * C;
#pragma unroll
  for (int nt = 0; nt < 4; ++nt)
#pragma unroll
    for (int r = 0; r < 8; ++r) {
      const int trow = q0 + r + laneHi * 8;
      const int col  = h * HD + nt * 16 + lane15;
      yo[(size_t)trow * C + col] = f32_to_bf16(y[nt][r] * lrow[r]);
    }
}

extern "C" void kernel_launch(void* const* d_in, const int* in_sizes, int n_in,
                              void* d_out, int out_size, void* d_ws, size_t ws_size,
                              hipStream_t stream) {
  (void)in_sizes; (void)n_in; (void)out_size; (void)ws_size;
  const float* x      = (const float*)d_in[0];
  const float* w_qkv  = (const float*)d_in[1];
  const float* w_proj = (const float*)d_in[2];
  float* out = (float*)d_out;

  const int M = BATCH * T;  // 8192
  unsigned short* qkvws = (unsigned short*)d_ws;                       // 48 MB
  unsigned short* yws   = qkvws + (size_t)3 * BATCH * NH * T * HD;     // 16 MB
  unsigned short* xbf   = yws + (size_t)M * C;                         // 16 MB
  unsigned short* wqkvT = xbf + (size_t)M * C;                         //  6 MB
  unsigned short* wprojT = wqkvT + (size_t)C * 3 * C;                  //  2 MB

  dim3 blk(256);

  // pre-passes: x -> bf16; weights -> bf16 transposed [N][K]
  convert_bf16_kernel<<<dim3((M * C) / (256 * 8)), blk, 0, stream>>>(x, xbf);
  transpose_bf16_kernel<<<dim3((C * 3 * C) / 256), blk, 0, stream>>>(w_qkv, wqkvT, 3 * C);
  transpose_bf16_kernel<<<dim3((C * C) / 256), blk, 0, stream>>>(w_proj, wprojT, C);

  // 1) qkv = x @ w_qkv  -> bf16 scatter [3][B][H][T][D]
  gemm_wmma_kernel<1><<<dim3((3 * C) / 128, M / 128), blk, 0, stream>>>(
      xbf, wqkvT, nullptr, qkvws, M, 3 * C, C);
  // 2) flash attention -> y bf16 [B][T][C]
  attn_wmma_kernel<<<dim3((BATCH * NH * (T / 16)) / 8), blk, 0, stream>>>(qkvws, yws);
  // 3) out = y @ w_proj -> f32 [B][T][C]
  gemm_wmma_kernel<0><<<dim3(C / 128, M / 128), blk, 0, stream>>>(
      yws, wprojT, out, nullptr, M, C, C);
}
